// Contrastive_Loss_32710470926820
// MI455X (gfx1250) — matrix operand order repaired and checked by reading, not compiled, not run on hardware
//
#include <hip/hip_runtime.h>
#include <hip/hip_bf16.h>

typedef __attribute__((ext_vector_type(16))) _Float16 v16h;
typedef __attribute__((ext_vector_type(8)))  float    v8f;
typedef __attribute__((ext_vector_type(8)))  unsigned v8u;
typedef __attribute__((ext_vector_type(4)))  unsigned v4u;
typedef __attribute__((ext_vector_type(8)))  int      v8i;
typedef __attribute__((ext_vector_type(4)))  int      v4i;

#define WAVES_PER_BLOCK 4
#define CONV_THREADS (WAVES_PER_BLOCK * 32)
#define KC 128          // K-chunk staged in LDS (f32, packed [16][kc])

#ifndef __has_builtin
#define __has_builtin(x) 0
#endif

#if __has_builtin(__builtin_amdgcn_tensor_load_to_lds) && \
    __has_builtin(__builtin_amdgcn_s_wait_tensorcnt)
#define HAVE_TDM 1
#else
#define HAVE_TDM 0
#endif

union H2u { unsigned u; _Float16 h[2]; };

// ---------------------------------------------------------------------------
// Implicit-GEMM conv via V_WMMA_F32_16X16X32_F16.
//   A (16xK)  : weights for 16 output channels; 16 x kc f32 slab staged into
//               LDS by the Tensor Data Mover (one TENSOR_LOAD_TO_LDS per
//               chunk, S_WAIT_TENSORCNT), converted to f16 pairs on fragment
//               build with v_cvt_pkrtz.
//   B (Kx16)  : im2col gather of 16 output pixels, built in registers.
//               Each wave owns TWO N-tiles sharing one A fragment
//               (2 WMMAs per K-step).
//   D (16x16) : f32 accumulators -> global (f32 raw, or f16 + fused bias).
// KSIZE==3: pad=1, stride s.  KSIZE==1: plain channel GEMM.
// ---------------------------------------------------------------------------
template <typename Tin, int KSIZE, bool FUSE_BIAS_F16>
__global__ __launch_bounds__(CONV_THREADS)
void conv_wmma_kernel(const Tin* __restrict__ in, const float* __restrict__ wflat,
                      const float* __restrict__ bias, float* __restrict__ outf,
                      _Float16* __restrict__ outh,
                      int Ci, int Co, int H, int W, int Ho, int Wo, int stride, int B)
{
    __shared__ float Asf[16 * KC];         // LDS offset 0 (only shared object)
    const int lane = threadIdx.x & 31;
    const int wave = threadIdx.x >> 5;
    const int K    = Ci * KSIZE * KSIZE;   // multiple of 32 for all shapes here
    const int HWo  = Ho * Wo;
    const int N    = B * HWo;              // multiple of 32 for all shapes here
    const int npairs = N >> 5;             // pairs of 16-wide N tiles

    int pair = blockIdx.x * WAVES_PER_BLOCK + wave;
    const bool nvalid = (pair < npairs);
    if (!nvalid) pair = npairs - 1;        // clamp: keep EXEC all-ones for WMMA
    const int tile_m = blockIdx.y;

    // Decode this lane's two output pixels (B-matrix columns of each tile).
    const int l16  = lane & 15;
    const int nb0  = pair * 32 + l16;
    const int nb1  = nb0 + 16;
    const int bb0  = nb0 / HWo,  ohw0 = nb0 - bb0 * HWo;
    const int bb1  = nb1 / HWo,  ohw1 = nb1 - bb1 * HWo;
    const int oh0  = ohw0 / Wo,  ow0 = ohw0 - oh0 * Wo;
    const int oh1  = ohw1 / Wo,  ow1 = ohw1 - oh1 * Wo;

    v8f acc0 = {}, acc1 = {};
    const float* wbase = wflat + (size_t)(tile_m * 16) * K;

    // Per-lane B-fragment builder (32x16 layout: lane%16 = N, lane/16 = K half)
    auto buildB = [&](int bb, int ohw, int oh, int ow, int kbb) -> v16h {
        v8u bu;
#pragma unroll
        for (int i = 0; i < 8; ++i) {
            H2u p;
#pragma unroll
            for (int e2 = 0; e2 < 2; ++e2) {
                int k = kbb + ((i < 4) ? 2 * i : 16 + 2 * (i - 4)) + e2;
                float v;
                if (KSIZE == 1) {
                    v = (float)in[((size_t)bb * Ci + k) * HWo + ohw];
                } else {
                    int ci = k / 9, rc = k - ci * 9;
                    int r = rc / 3, c = rc - r * 3;
                    int ih = oh * stride + r - 1;
                    int iw = ow * stride + c - 1;
                    v = (ih >= 0 && ih < H && iw >= 0 && iw < W)
                            ? (float)in[(((size_t)bb * Ci + ci) * H + ih) * W + iw]
                            : 0.0f;
                }
                p.h[e2] = (_Float16)v;
            }
            bu[i] = p.u;
        }
        return __builtin_bit_cast(v16h, bu);
    };

    for (int k0 = 0; k0 < K; k0 += KC) {
        int kc = K - k0; if (kc > KC) kc = KC;     // remainder stays mult. of 32
        __syncthreads();                           // previous chunk fully read

#if HAVE_TDM
        // --- Tensor Data Mover: DMA the 16 x kc f32 weight slab into LDS ---
        if (wave == 0) {
            unsigned long long ga = (unsigned long long)(uintptr_t)(wbase + k0);
            v4u g0;
            g0[0] = 1u;                                   // count=1 (valid D#)
            g0[1] = 0u;                                   // lds_addr = 0 (Asf)
            g0[2] = (unsigned)(ga & 0xffffffffu);         // global_addr lo
            g0[3] = (unsigned)((ga >> 32) & 0x1ffffffu)   // global_addr hi
                    | (2u << 30);                         // type = 2 ("image")
            v8i g1;
            g1[0] = 0x20000;                              // data_size = 4 bytes
            g1[1] = (int)(((unsigned)K & 0xffffu) << 16); // tensor_dim0[15:0]
            g1[2] = (int)(((unsigned)K >> 16) | (16u << 16)); // dim0 hi, dim1=16
            g1[3] = (int)((unsigned)kc << 16);            // tile_dim0 = kc
            g1[4] = 16;                                   // tile_dim1 = 16
            g1[5] = K;                                    // tensor_dim0_stride
            g1[6] = 0;
            g1[7] = 0;
            v4i gz4 = {};
#if __clang_major__ >= 23
            v8i gz8 = {};
            __builtin_amdgcn_tensor_load_to_lds(g0, g1, gz4, gz4, gz8, 0);
#else
            __builtin_amdgcn_tensor_load_to_lds(g0, g1, gz4, gz4, 0);
#endif
            __builtin_amdgcn_s_wait_tensorcnt(0);
        }
#else
        // --- Fallback: cooperative stage of the weight slab into LDS ---
        for (int e = threadIdx.x; e < 16 * kc; e += CONV_THREADS) {
            int m = e / kc, kk = e - m * kc;
            Asf[e] = wbase[(size_t)m * K + k0 + kk];
        }
#endif
        if (k0 + KC < K)
            __builtin_prefetch(wbase + k0 + KC, 0, 1);    // global_prefetch_b8
        __syncthreads();                                  // slab visible to all

        const int steps = kc >> 5;
        for (int st = 0; st < steps; ++st) {
            const int ks = st * 32;

            // ---- A fragment (ISA 16-bit 16x32 layout) from f32 LDS ----
            v8u au;
            const int am     = lane & 15;
            const int kbaseA = (lane < 16) ? 0 : 8;
#pragma unroll
            for (int i = 0; i < 8; ++i) {
                int kl  = kbaseA + ((i < 4) ? 2 * i : 16 + 2 * (i - 4));
                int idx = am * kc + ks + kl;
                float f0 = Asf[idx];
                float f1 = Asf[idx + 1];
                au[i] = __builtin_bit_cast(unsigned,
                            __builtin_amdgcn_cvt_pkrtz(f0, f1));
            }
            v16h a = __builtin_bit_cast(v16h, au);

            // ---- Two B fragments, one shared A: 2 WMMAs per K-step ----
            const int kbb = k0 + ks + ((lane < 16) ? 0 : 8);
            v16h b0 = buildB(bb0, ohw0, oh0, ow0, kbb);
            acc0 = __builtin_amdgcn_wmma_f32_16x16x32_f16(
                       false, a, false, b0, (short)0, acc0, false, false);
            v16h b1 = buildB(bb1, ohw1, oh1, ow1, kbb);
            acc1 = __builtin_amdgcn_wmma_f32_16x16x32_f16(
                       false, a, false, b1, (short)0, acc1, false, false);
        }
    }

    if (nvalid) {
#pragma unroll
        for (int j = 0; j < 8; ++j) {
            int m  = (lane < 16) ? j : (8 + j);        // C/D layout per ISA
            int co = tile_m * 16 + m;
            size_t o0 = ((size_t)bb0 * Co + co) * HWo + ohw0;
            size_t o1 = ((size_t)bb1 * Co + co) * HWo + ohw1;
            if constexpr (FUSE_BIAS_F16) {
                float bs = bias[co];
                outh[o0] = (_Float16)(acc0[j] + bs);
                outh[o1] = (_Float16)(acc1[j] + bs);
            } else {
                outf[o0] = acc0[j];
                outf[o1] = acc1[j];
            }
        }
    }
}

// ---------------------------------------------------------------------------
// Per-channel training-mode BN statistics: mean and 1/sqrt(var+eps).
// ---------------------------------------------------------------------------
__global__ __launch_bounds__(256)
void bn_stats_kernel(const float* __restrict__ x, float* __restrict__ stats,
                     int Co, int HW, int B)
{
    __shared__ float s1[256], s2[256];
    const int c = blockIdx.x;
    float sum = 0.f, sq = 0.f;
    for (int b = 0; b < B; ++b) {
        const float* p = x + ((size_t)b * Co + c) * HW;
        for (int i = threadIdx.x; i < HW; i += 256) {
            float v = p[i]; sum += v; sq += v * v;
        }
    }
    s1[threadIdx.x] = sum; s2[threadIdx.x] = sq;
    __syncthreads();
    for (int o = 128; o > 0; o >>= 1) {
        if (threadIdx.x < o) {
            s1[threadIdx.x] += s1[threadIdx.x + o];
            s2[threadIdx.x] += s2[threadIdx.x + o];
        }
        __syncthreads();
    }
    if (threadIdx.x == 0) {
        float n = (float)(B * HW);
        float m = s1[0] / n;
        float var = s2[0] / n - m * m;
        stats[2 * c]     = m;
        stats[2 * c + 1] = rsqrtf(var + 1e-5f);
    }
}

// ---------------------------------------------------------------------------
// BN apply + LeakyReLU(0.1), writes f16 activation for the next WMMA conv.
// ---------------------------------------------------------------------------
__global__ void bn_lrelu_kernel(const float* __restrict__ x, const float* __restrict__ stats,
                                const float* __restrict__ g, const float* __restrict__ gb,
                                _Float16* __restrict__ y, int Co, int HW, int total)
{
    int idx = blockIdx.x * blockDim.x + threadIdx.x;
    if (idx >= total) return;
    int c = (idx / HW) % Co;
    float v = (x[idx] - stats[2 * c]) * stats[2 * c + 1] * g[c] + gb[c];
    v = (v >= 0.f) ? v : 0.1f * v;
    y[idx] = (_Float16)v;
}

// ---------------------------------------------------------------------------
// Global average pool over HW for [B=2, 256, HW] f16 -> fea [B,256] f32.
// ---------------------------------------------------------------------------
__global__ __launch_bounds__(64)
void pool_kernel(const _Float16* __restrict__ act, float* __restrict__ fea, int HW)
{
    __shared__ float s[64];
    const int b = blockIdx.x >> 8;
    const int c = blockIdx.x & 255;
    const _Float16* p = act + ((size_t)(b * 256 + c)) * HW;
    float sum = 0.f;
    for (int i = threadIdx.x; i < HW; i += 64) sum += (float)p[i];
    s[threadIdx.x] = sum;
    __syncthreads();
    for (int o = 32; o > 0; o >>= 1) {
        if (threadIdx.x < o) s[threadIdx.x] += s[threadIdx.x + o];
        __syncthreads();
    }
    if (threadIdx.x == 0) fea[b * 256 + c] = s[0] / (float)HW;
}

// ---------------------------------------------------------------------------
// 2-layer MLP head: emb = lrelu(fea @ w1 + b1) @ w2 + b2   (per sample)
// ---------------------------------------------------------------------------
__global__ __launch_bounds__(256)
void fc_kernel(const float* __restrict__ fea,
               const float* __restrict__ w1, const float* __restrict__ b1,
               const float* __restrict__ w2, const float* __restrict__ b2,
               float* __restrict__ emb)
{
    __shared__ float sf[256], sh[256];
    const int b = blockIdx.x, t = threadIdx.x;
    sf[t] = fea[b * 256 + t];
    __syncthreads();
    float acc = b1[t];
    for (int i = 0; i < 256; ++i) acc += sf[i] * w1[i * 256 + t];
    acc = (acc >= 0.f) ? acc : 0.1f * acc;
    sh[t] = acc;
    __syncthreads();
    float o = b2[t];
    for (int i = 0; i < 256; ++i) o += sh[i] * w2[i * 256 + t];
    emb[b * 256 + t] = o;
}

// ---------------------------------------------------------------------------
// Contrastive losses on [S=4, B=2, 256] embeddings.
// ---------------------------------------------------------------------------
__device__ float one_side_loss(const float (*GPP)[8], const float (*GPQ)[8])
{
    const int S = 4, Bn = 2;
    float loss = 0.f;
    for (int i = 0; i < S; ++i)
        for (int b = 0; b < Bn; ++b) {
            int ib = i * Bn + b;
            float sA = 0.f, sAQ = 0.f, sameA = 0.f, sameAQ = 0.f;
            for (int l = 0; l < S; ++l)
                for (int m = 0; m < Bn; ++m) {
                    sA  += __expf(GPP[ib][l * Bn + m]);
                    sAQ += __expf(GPQ[ib][l * Bn + m]);
                }
            for (int m = 0; m < Bn; ++m) {
                sameA  += __expf(GPP[ib][i * Bn + m]);
                sameAQ += __expf(GPQ[ib][i * Bn + m]);
            }
            float den = sA + sAQ - sameA - sameAQ;
            float t1 = 0.f, c = 0.f;
            for (int m = 0; m < Bn; ++m) {
                if (m != b) {
                    c += __expf(GPP[ib][i * Bn + m]);   // cumulative numerator
                    t1 += -__logf(c / (den + c));
                }
            }
            float t2 = 0.f;
            for (int m = 0; m < Bn; ++m) {
                float q = __expf(GPQ[ib][i * Bn + m]);
                t2 += -__logf(q / (den + q));
            }
            loss += (t1 + t2) / (2.0f * Bn - 1.0f);
        }
    return loss;
}

__global__ __launch_bounds__(256)
void loss_kernel(const float* __restrict__ L, const float* __restrict__ Lb,
                 const float* __restrict__ Sm, float* __restrict__ out)
{
    __shared__ float tab[4][8][8];   // 0:L.L  1:Lb.Lb  2:L.Lb  3:Sem.Sem (scaled 1/TEMP)
    const int t = threadIdx.x;
    const int tb = t >> 6, e = t & 63, v1 = e >> 3, v2 = e & 7;
    const float* X = (tb == 1) ? Lb : (tb == 3) ? Sm : L;
    const float* Y = (tb == 0) ? L : (tb == 3) ? Sm : Lb;
    float d = 0.f;
    for (int i = 0; i < 256; ++i) d += X[v1 * 256 + i] * Y[v2 * 256 + i];
    tab[tb][v1][v2] = d * 10.0f;     // / TEMP (0.1)
    __syncthreads();

    if (t == 0) {
        float T[8][8];
        for (int a = 0; a < 8; ++a)
            for (int b = 0; b < 8; ++b) T[a][b] = tab[2][b][a];  // Lb.L = (L.Lb)^T
        float loc = one_side_loss(tab[0], tab[2]) + one_side_loss(tab[1], T);

        const int S = 4, Bn = 2;
        float sem = 0.f;
        for (int i = 0; i < S; ++i)
            for (int j = 0; j < Bn; ++j) {
                int ij = i * Bn + j;
                float sum = 0.f, diagsum = 0.f;
                for (int l = 0; l < S; ++l)
                    for (int m = 0; m < Bn; ++m) sum += __expf(tab[3][ij][l * Bn + m]);
                for (int l = 0; l < S; ++l) diagsum += __expf(tab[3][ij][l * Bn + j]);
                float den = sum - diagsum;
                float tt = 0.f;
                for (int l = 0; l < S; ++l) {
                    if (l != i) {
                        float dd = __expf(tab[3][ij][l * Bn + j]);
                        tt += -__logf(dd / (den + dd));
                    }
                }
                sem += tt / (float)(S - 1);
            }
        out[0] = loc;
        out[1] = sem;
        out[2] = 0.f;
    }
}

// ---------------------------------------------------------------------------
// Host orchestration.
// Input flatten order (pytree, dict keys sorted):
//   0..3 xb0..xb3, 4..7 x0..x3,
//   loc_params @8 : b1,b2, cb[0..5], cw[0..5], g[0..5], gb[0..5], w1,w2
//   sem_params @36: same layout
//   ct_params @64 : b[0..3] then w[0..3]
// ---------------------------------------------------------------------------
extern "C" void kernel_launch(void* const* d_in, const int* in_sizes, int n_in,
                              void* d_out, int out_size, void* d_ws, size_t ws_size,
                              hipStream_t stream)
{
    (void)in_sizes; (void)n_in; (void)out_size; (void)ws_size;

    static const int CHS_I[6] = {256, 64, 64, 128, 128, 256};
    static const int CHS_O[6] = {64, 64, 128, 128, 256, 256};
    static const int STR[6]   = {1, 1, 2, 1, 2, 1};
    static const int Hs[4]    = {128, 64, 32, 16};
    const int B = 2;

    auto P = [&](int i) { return (const float*)d_in[i]; };

    // Workspace carve-up (256B aligned).
    char*  ws  = (char*)d_ws;
    size_t off = 0;
    auto alloc = [&](size_t bytes) {
        void* p = ws + off;
        off += (bytes + 255) & ~(size_t)255;
        return p;
    };
    _Float16* actA  = (_Float16*)alloc((size_t)2 * 256 * 128 * 128 * 2); // big act (temp_b)
    float*    convo = (float*)   alloc((size_t)2 * 64 * 128 * 128 * 4);  // conv f32 out
    _Float16* actC  = (_Float16*)alloc((size_t)2 * 64 * 128 * 128 * 2);  // small act
    float*    stats = (float*)alloc(512 * sizeof(float));
    float*    fea   = (float*)alloc(512 * sizeof(float));
    float*    Lb    = (float*)alloc(4 * 2 * 256 * sizeof(float));
    float*    L     = (float*)alloc(4 * 2 * 256 * sizeof(float));
    float*    Sem   = (float*)alloc(4 * 2 * 256 * sizeof(float));

    auto encoder = [&](const void* in0, bool in_f16, int H0, int pb, float* embRow) {
        const void* cur = in0;
        int H = H0;
        for (int i = 0; i < 6; ++i) {
            const int Ci = CHS_I[i], Co = CHS_O[i], s = STR[i];
            const int Ho = (H + s - 1) / s;
            const int HWo = Ho * Ho;
            const int N = B * HWo;
            dim3 grid(((N >> 5) + WAVES_PER_BLOCK - 1) / WAVES_PER_BLOCK, Co / 16);
            const float* w = P(pb + 8 + i);
            if (i == 0 && !in_f16)
                conv_wmma_kernel<float, 3, false><<<grid, CONV_THREADS, 0, stream>>>(
                    (const float*)cur, w, nullptr, convo, nullptr,
                    Ci, Co, H, H, Ho, Ho, s, B);
            else
                conv_wmma_kernel<_Float16, 3, false><<<grid, CONV_THREADS, 0, stream>>>(
                    (const _Float16*)cur, w, nullptr, convo, nullptr,
                    Ci, Co, H, H, Ho, Ho, s, B);

            bn_stats_kernel<<<Co, 256, 0, stream>>>(convo, stats, Co, HWo, B);

            const int total = B * Co * HWo;
            _Float16* oact = (i & 1) ? actA : actC;
            bn_lrelu_kernel<<<(total + 255) / 256, 256, 0, stream>>>(
                convo, stats, P(pb + 14 + i), P(pb + 20 + i), oact, Co, HWo, total);

            cur = oact;
            H = Ho;
        }
        const int HWf = H * H;   // final layer output always 256 channels
        pool_kernel<<<B * 256, 64, 0, stream>>>((const _Float16*)cur, fea, HWf);
        fc_kernel<<<B, 256, 0, stream>>>(fea, P(pb + 26), P(pb + 0),
                                         P(pb + 27), P(pb + 1), embRow);
    };

    // Path 1: temp_b = conv1x1(xb_i) -> encoder(loc) -> Lb
    for (int i = 0; i < 4; ++i) {
        const int H = Hs[i], HW = H * H, Ci = 256 << i, N = B * HW;
        dim3 grid(((N >> 5) + WAVES_PER_BLOCK - 1) / WAVES_PER_BLOCK, 256 / 16);
        conv_wmma_kernel<float, 1, true><<<grid, CONV_THREADS, 0, stream>>>(
            (const float*)d_in[i], P(68 + i), P(64 + i), nullptr, actA,
            Ci, 256, H, H, H, H, 1, B);
        encoder(actA, true, H, 8, Lb + i * B * 256);
    }
    // Path 2: x_i -> encoder(loc) -> L
    for (int i = 0; i < 4; ++i)
        encoder(d_in[4 + i], false, Hs[i], 8, L + i * B * 256);
    // Path 3: x_i -> encoder(sem) -> Sem
    for (int i = 0; i < 4; ++i)
        encoder(d_in[4 + i], false, Hs[i], 36, Sem + i * B * 256);

    loss_kernel<<<1, 256, 0, stream>>>(L, Lb, Sem, (float*)d_out);
}
